// FastRCNNRelationFCHead_88390426952099
// MI455X (gfx1250) — matrix operand order repaired and checked by reading, not compiled, not run on hardware
//
#include <hip/hip_runtime.h>

// ---------------- types ----------------
typedef __attribute__((ext_vector_type(16))) __bf16 v16bf;
typedef __attribute__((ext_vector_type(8)))  __bf16 v8bf;
typedef __attribute__((ext_vector_type(4)))  __bf16 v4bf;
typedef __attribute__((ext_vector_type(8)))  float  v8f;

union BF16x16 { v16bf v; v8bf h[2]; };

#define WMMA_BF16(a, b, c) \
  __builtin_amdgcn_wmma_f32_16x16x32_bf16(false, (a), false, (b), (short)0, (c), false, false)

// A-operand 16x32 bf16 fragment: lane half h -> elems 0..7 = K(8h..8h+7), elems 8..15 = K(16+8h..)
__device__ __forceinline__ v16bf make_fragA(const __bf16* base, int half) {
  BF16x16 u;
  u.h[0] = *(const v8bf*)(base + 8 * half);
  u.h[1] = *(const v8bf*)(base + 16 + 8 * half);
  return u.v;
}
// B-operand 32x16 bf16 fragment (from Bt = N x K row-major): lane half h -> elems 0..7 = K(8h..8h+7), elems 8..15 = K(16+8h..)
__device__ __forceinline__ v16bf make_fragB(const __bf16* base, int half) {
  BF16x16 u;
  u.h[0] = *(const v8bf*)(base + 8 * half);
  u.h[1] = *(const v8bf*)(base + 16 + 8 * half);
  return u.v;
}

// CDNA5 async global->LDS copy (ASYNCcnt-tracked, no VGPR staging)
__device__ __forceinline__ void async_copy_b128(unsigned lds_off, unsigned long long gaddr) {
  asm volatile("global_load_async_to_lds_b128 %0, %1, off"
               :: "v"(lds_off), "v"(gaddr) : "memory");
}
__device__ __forceinline__ void wait_async0() {
  asm volatile("s_wait_asynccnt 0x0" ::: "memory");
}

// ---------------- elementwise f32 -> bf16 ----------------
__global__ void cvt_f32_bf16(const float* __restrict__ in, __bf16* __restrict__ out, long n) {
  long i = ((long)blockIdx.x * 256 + threadIdx.x) * 4;
  if (i + 3 < n) {
    float4 f = *(const float4*)(in + i);
    v4bf o; o.x = (__bf16)f.x; o.y = (__bf16)f.y; o.z = (__bf16)f.z; o.w = (__bf16)f.w;
    *(v4bf*)(out + i) = o;
  } else {
    for (; i < n; ++i) out[i] = (__bf16)in[i];
  }
}

// ---------------- transpose + convert: in (K x N) f32 -> out (N x K) bf16 ----------------
__global__ void transpose_cvt(const float* __restrict__ in, __bf16* __restrict__ out,
                              int K, int N) {
  __shared__ float tile[32][33];
  int k0 = blockIdx.x * 32, n0 = blockIdx.y * 32;
  int tx = threadIdx.x, ty = threadIdx.y;        // block (32, 8)
  for (int i = ty; i < 32; i += 8)
    if (k0 + i < K && n0 + tx < N) tile[i][tx] = in[(size_t)(k0 + i) * N + n0 + tx];
  __syncthreads();
  for (int i = ty; i < 32; i += 8)
    if (n0 + i < N && k0 + tx < K) out[(size_t)(n0 + i) * K + k0 + tx] = (__bf16)tile[tx][i];
}

// ---------------- bf16 WMMA GEMM: C(MxN) = A(MxK) * Bt(NxK)^T (+bias) ----------------
#define BM 128
#define BN 128
#define BKK 64
#define LDK 72

__global__ void __launch_bounds__(256, 1)
gemm_bf16(const __bf16* __restrict__ A, const __bf16* __restrict__ Bt,
          const float* __restrict__ bias,
          float* outF, __bf16* outB, __bf16* outT,
          int M, int N, int K, int lda, int ldb, int ldc, int ldT,
          long strideB, long strideT) {
  __shared__ __bf16 sA[BM * LDK];
  __shared__ __bf16 sB[BN * LDK];
  int t = threadIdx.x;
  int l = t & 31, wid = t >> 5;
  int half = l >> 4, lc = l & 15;
  int wm = (wid & 3) * 32;   // 4 waves along M
  int wn = (wid >> 2) * 64;  // 2 waves along N
  int tm = blockIdx.x * BM, tn = blockIdx.y * BN;
  int z = blockIdx.z;
  Bt += (size_t)z * strideB;

  // Per-thread tile-load slots: rows clamped (out-of-range rows only feed
  // store-guarded C entries), so the hot loop has no divergence.
  unsigned long long gA[4], gB[4];
  unsigned ldsAoff[4], ldsBoff[4];
#pragma unroll
  for (int i = 0; i < 4; ++i) {
    int lin = t + i * 256;
    int r = lin >> 3, kk = (lin & 7) * 8;
    int ra = tm + r; if (ra > M - 1) ra = M - 1;
    int rb = tn + r; if (rb > N - 1) rb = N - 1;
    gA[i] = (unsigned long long)(uintptr_t)(A  + (size_t)ra * lda + kk);
    gB[i] = (unsigned long long)(uintptr_t)(Bt + (size_t)rb * ldb + kk);
    ldsAoff[i] = (unsigned)(uintptr_t)(void*)(sA + r * LDK + kk);
    ldsBoff[i] = (unsigned)(uintptr_t)(void*)(sB + r * LDK + kk);
  }

  v8f acc[2][4];
#pragma unroll
  for (int i = 0; i < 2; ++i)
#pragma unroll
    for (int j = 0; j < 4; ++j) acc[i][j] = (v8f)(0.0f);

  for (int k0 = 0; k0 < K; k0 += BKK) {
#pragma unroll
    for (int i = 0; i < 4; ++i) {
      async_copy_b128(ldsAoff[i], gA[i]);
      async_copy_b128(ldsBoff[i], gB[i]);
      gA[i] += 2ull * BKK;
      gB[i] += 2ull * BKK;
    }
    // prefetch next k-slice into L2 (global_prefetch_b8)
    if (k0 + BKK < K) {
      __builtin_prefetch((const void*)(uintptr_t)gA[0], 0, 1);
      __builtin_prefetch((const void*)(uintptr_t)gB[0], 0, 1);
    }
    wait_async0();
    __syncthreads();
#pragma unroll
    for (int ks = 0; ks < BKK; ks += 32) {
      v16bf a0 = make_fragA(sA + (wm + lc) * LDK + ks, half);
      v16bf a1 = make_fragA(sA + (wm + 16 + lc) * LDK + ks, half);
      v16bf b0 = make_fragB(sB + (wn + lc) * LDK + ks, half);
      v16bf b1 = make_fragB(sB + (wn + 16 + lc) * LDK + ks, half);
      v16bf b2 = make_fragB(sB + (wn + 32 + lc) * LDK + ks, half);
      v16bf b3 = make_fragB(sB + (wn + 48 + lc) * LDK + ks, half);
      acc[0][0] = WMMA_BF16(a0, b0, acc[0][0]);
      acc[0][1] = WMMA_BF16(a0, b1, acc[0][1]);
      acc[0][2] = WMMA_BF16(a0, b2, acc[0][2]);
      acc[0][3] = WMMA_BF16(a0, b3, acc[0][3]);
      acc[1][0] = WMMA_BF16(a1, b0, acc[1][0]);
      acc[1][1] = WMMA_BF16(a1, b1, acc[1][1]);
      acc[1][2] = WMMA_BF16(a1, b2, acc[1][2]);
      acc[1][3] = WMMA_BF16(a1, b3, acc[1][3]);
    }
    __syncthreads();
  }

#pragma unroll
  for (int i = 0; i < 2; ++i)
#pragma unroll
    for (int j = 0; j < 4; ++j)
#pragma unroll
      for (int vr = 0; vr < 8; ++vr) {
        int r = tm + wm + 16 * i + vr + 8 * half;
        int c = tn + wn + 16 * j + lc;
        if (r < M && c < N) {
          float v = acc[i][j][vr] + (bias ? bias[c] : 0.0f);
          if (outF) outF[(size_t)r * ldc + c] = v;
          if (outB) outB[(size_t)r * ldc + c] = (__bf16)v;
          if (outT) outT[(size_t)z * strideT + (size_t)c * ldT + r] = (__bf16)v;
        }
      }
}

// ---------------- position embedding -> log(relu(emb@Wpos+b)) : aff[g][n][m] ----------------
__global__ void poslog_kernel(const float* __restrict__ bbox, const float* __restrict__ Wpos,
                              const float* __restrict__ bpos, float* __restrict__ aff) {
  __shared__ float sW[64 * 16];
  __shared__ float sb[16];
  for (int i = threadIdx.x; i < 1024; i += 256) sW[i] = Wpos[i];
  if (threadIdx.x < 16) sb[threadIdx.x] = bpos[threadIdx.x];
  __syncthreads();
  long idx = (long)blockIdx.x * 256 + threadIdx.x;
  if (idx >= 4096L * 300L) return;
  int n = (int)(idx / 300), m = (int)(idx % 300);
  float xn0 = bbox[n * 4 + 0], yn0 = bbox[n * 4 + 1], xn1 = bbox[n * 4 + 2], yn1 = bbox[n * 4 + 3];
  float xm0 = bbox[m * 4 + 0], ym0 = bbox[m * 4 + 1], xm1 = bbox[m * 4 + 2], ym1 = bbox[m * 4 + 3];
  float bwn = xn1 - xn0 + 1.0f, bhn = yn1 - yn0 + 1.0f;
  float bwm = xm1 - xm0 + 1.0f, bhm = ym1 - ym0 + 1.0f;
  float cxn = 0.5f * (xn0 + xn1), cyn = 0.5f * (yn0 + yn1);
  float cxm = 0.5f * (xm0 + xm1), cym = 0.5f * (ym0 + ym1);
  float p[4];
  p[0] = logf(fmaxf(fabsf((cxn - cxm) / bwn), 1e-3f));
  p[1] = logf(fmaxf(fabsf((cyn - cym) / bhn), 1e-3f));
  p[2] = logf(bwn / bwm);
  p[3] = logf(bhn / bhm);
  const float sc[8] = {100.0f, 42.1696503f, 17.7827941f, 7.49894209f,
                       3.16227766f, 1.33352143f, 0.562341325f, 0.237137371f};
  float acc[16];
#pragma unroll
  for (int g = 0; g < 16; ++g) acc[g] = 0.0f;
#pragma unroll
  for (int c = 0; c < 4; ++c)
#pragma unroll
    for (int tt = 0; tt < 8; ++tt) {
      float d = p[c] * sc[tt];
      float sn, cs;
      __sincosf(d, &sn, &cs);
      const float* ws = &sW[(c * 16 + tt) * 16];
      const float* wc = &sW[(c * 16 + 8 + tt) * 16];
#pragma unroll
      for (int g = 0; g < 16; ++g) acc[g] += sn * ws[g] + cs * wc[g];
    }
#pragma unroll
  for (int g = 0; g < 16; ++g) {
    float v = fmaxf(acc[g] + sb[g], 1e-6f);
    aff[((size_t)g * 4096 + n) * 320 + m] = logf(v);
  }
}

// ---------------- flash attention: one wave per (16-row n-tile, group) ----------------
__global__ void __launch_bounds__(32, 1)
attn_kernel(const __bf16* __restrict__ q, const __bf16* __restrict__ kmat,
            const __bf16* __restrict__ kwt, const float* __restrict__ aff,
            const float* __restrict__ xfc, const float* __restrict__ bo,
            float* outF, __bf16* outB) {
  __shared__ __bf16 pT[16 * 32];
  int l = threadIdx.x;  // block = 32 (one wave)
  int half = l >> 4, lc = l & 15;
  int g = blockIdx.y;
  int nt = blockIdx.x * 16;

  v16bf qf0 = make_fragA(q + (size_t)(nt + lc) * 1024 + g * 64 + 0, half);
  v16bf qf1 = make_fragA(q + (size_t)(nt + lc) * 1024 + g * 64 + 32, half);

  v8f acc[4];
#pragma unroll
  for (int j = 0; j < 4; ++j) acc[j] = (v8f)(0.0f);
  float mrun[8], lrun[8];
#pragma unroll
  for (int r = 0; r < 8; ++r) { mrun[r] = -1e30f; lrun[r] = 0.0f; }

  const float* affb = aff + ((size_t)g * 4096 + nt) * 320;

  for (int mt = 0; mt < 320; mt += 32) {
    v8f s0 = (v8f)(0.0f), s1 = (v8f)(0.0f);
    {
      v16bf k00 = make_fragB(kmat + (size_t)(mt + lc) * 1024 + g * 64, half);
      v16bf k10 = make_fragB(kmat + (size_t)(mt + 16 + lc) * 1024 + g * 64, half);
      s0 = WMMA_BF16(qf0, k00, s0);
      s1 = WMMA_BF16(qf0, k10, s1);
      v16bf k01 = make_fragB(kmat + (size_t)(mt + lc) * 1024 + g * 64 + 32, half);
      v16bf k11 = make_fragB(kmat + (size_t)(mt + 16 + lc) * 1024 + g * 64 + 32, half);
      s0 = WMMA_BF16(qf1, k01, s0);
      s1 = WMMA_BF16(qf1, k11, s1);
    }
    int m0 = mt + lc, m1 = mt + 16 + lc;
    __syncthreads();  // previous P tile fully consumed
#pragma unroll
    for (int vr = 0; vr < 8; ++vr) {
      int rr = vr + 8 * half;
      float v0 = (m0 < 300) ? s0[vr] * 0.125f + affb[rr * 320 + m0] : -1e30f;
      float v1 = (m1 < 300) ? s1[vr] * 0.125f + affb[rr * 320 + m1] : -1e30f;
      float cm = fmaxf(v0, v1);
#pragma unroll
      for (int msk = 1; msk < 16; msk <<= 1) cm = fmaxf(cm, __shfl_xor(cm, msk, 32));
      float mnew = fmaxf(mrun[vr], cm);
      float corr = __expf(mrun[vr] - mnew);
      float p0 = __expf(v0 - mnew);
      float p1 = __expf(v1 - mnew);
      float rs = p0 + p1;
#pragma unroll
      for (int msk = 1; msk < 16; msk <<= 1) rs += __shfl_xor(rs, msk, 32);
      lrun[vr] = lrun[vr] * corr + rs;
      mrun[vr] = mnew;
#pragma unroll
      for (int j = 0; j < 4; ++j) acc[j][vr] *= corr;
      pT[rr * 32 + lc] = (__bf16)p0;
      pT[rr * 32 + 16 + lc] = (__bf16)p1;
    }
    __syncthreads();
    v16bf pf = make_fragA(pT + lc * 32, half);
#pragma unroll
    for (int j = 0; j < 4; ++j) {
      v16bf kwf = make_fragB(kwt + ((size_t)g * 64 + 16 * j + lc) * 320 + mt, half);
      acc[j] = WMMA_BF16(pf, kwf, acc[j]);
    }
  }
#pragma unroll
  for (int j = 0; j < 4; ++j)
#pragma unroll
    for (int vr = 0; vr < 8; ++vr) {
      int n = nt + vr + 8 * half;
      int c = g * 64 + 16 * j + lc;
      float v = acc[j][vr] / lrun[vr] + bo[c] + xfc[(size_t)n * 1024 + c];
      v = fmaxf(v, 0.0f);
      if (outF) outF[(size_t)n * 1024 + c] = v;
      if (outB) outB[(size_t)n * 1024 + c] = (__bf16)v;
    }
}

// ---------------- host orchestration ----------------
extern "C" void kernel_launch(void* const* d_in, const int* in_sizes, int n_in,
                              void* d_out, int out_size, void* d_ws, size_t ws_size,
                              hipStream_t stream) {
  (void)in_sizes; (void)n_in; (void)out_size; (void)ws_size;
  const float* roi   = (const float*)d_in[0];
  const float* bbox  = (const float*)d_in[1];
  const float* Wfc1  = (const float*)d_in[2];
  const float* bfc1  = (const float*)d_in[3];
  const float* Wfc2  = (const float*)d_in[4];
  const float* bfc2  = (const float*)d_in[5];
  const float* Wpos[2] = {(const float*)d_in[6],  (const float*)d_in[14]};
  const float* bpos[2] = {(const float*)d_in[7],  (const float*)d_in[15]};
  const float* Wq[2]   = {(const float*)d_in[8],  (const float*)d_in[16]};
  const float* bq[2]   = {(const float*)d_in[9],  (const float*)d_in[17]};
  const float* Wk[2]   = {(const float*)d_in[10], (const float*)d_in[18]};
  const float* bk[2]   = {(const float*)d_in[11], (const float*)d_in[19]};
  const float* Wo[2]   = {(const float*)d_in[12], (const float*)d_in[20]};
  const float* bo[2]   = {(const float*)d_in[13], (const float*)d_in[21]};

  char* ws = (char*)d_ws;
  auto alloc = [&](size_t bytes) -> void* {
    void* p = (void*)ws;
    ws += (bytes + 255) & ~(size_t)255;
    return p;
  };
  __bf16* bfA  = (__bf16*)alloc(4096ull * 12544 * 2);  // roi bf16
  __bf16* bfW  = (__bf16*)alloc(1024ull * 12544 * 2);  // transposed weight bf16 (reused)
  __bf16* actA = (__bf16*)alloc(4096ull * 1024 * 2);   // activation bf16 ping
  __bf16* actB = (__bf16*)alloc(4096ull * 1024 * 2);   // activation bf16 pong
  __bf16* qbf  = (__bf16*)alloc(4096ull * 1024 * 2);
  __bf16* kbf  = (__bf16*)alloc(320ull * 1024 * 2);
  __bf16* wobf = (__bf16*)alloc(16ull * 64 * 1024 * 2);
  __bf16* kwt  = (__bf16*)alloc(16ull * 64 * 320 * 2); // KW^T per group: [g][o][m], ld 320
  float*  xfc  = (float*)alloc(4096ull * 1024 * 4);    // fc output f32 (residual)
  float*  aff  = (float*)alloc(16ull * 4096 * 320 * 4);

  dim3 tb32x8(32, 8);

  // fc1: x1 = roi @ Wfc1 + b
  cvt_f32_bf16<<<(51380224 / 4 + 255) / 256, 256, 0, stream>>>(roi, bfA, 51380224L);
  transpose_cvt<<<dim3(12544 / 32, 1024 / 32), tb32x8, 0, stream>>>(Wfc1, bfW, 12544, 1024);
  gemm_bf16<<<dim3(32, 8, 1), 256, 0, stream>>>(bfA, bfW, bfc1, xfc, actA, nullptr,
      4096, 1024, 12544, 12544, 12544, 1024, 0, 0L, 0L);

  __bf16* xin = actA;   // bf16 of current fc output
  for (int li = 0; li < 2; ++li) {
    // q = x @ Wq + bq  (bf16 out)
    transpose_cvt<<<dim3(32, 32), tb32x8, 0, stream>>>(Wq[li], bfW, 1024, 1024);
    gemm_bf16<<<dim3(32, 8, 1), 256, 0, stream>>>(xin, bfW, bq[li], nullptr, qbf, nullptr,
        4096, 1024, 1024, 1024, 1024, 1024, 0, 0L, 0L);
    // k = x[:300] @ Wk + bk  (bf16 out, 320-row padded buffer)
    transpose_cvt<<<dim3(32, 32), tb32x8, 0, stream>>>(Wk[li], bfW, 1024, 1024);
    gemm_bf16<<<dim3(3, 8, 1), 256, 0, stream>>>(xin, bfW, bk[li], nullptr, kbf, nullptr,
        300, 1024, 1024, 1024, 1024, 1024, 0, 0L, 0L);
    // KW^T[g] = (x[:300] @ Wo[g]^T)^T  -> kwt[g][o][m]
    cvt_f32_bf16<<<(1048576 / 4 + 255) / 256, 256, 0, stream>>>(Wo[li], wobf, 1048576L);
    gemm_bf16<<<dim3(3, 1, 16), 256, 0, stream>>>(xin, wobf, nullptr, nullptr, nullptr, kwt,
        300, 64, 1024, 1024, 1024, 64, 320, 65536L, 20480L);
    // position logits
    poslog_kernel<<<(4096 * 300 + 255) / 256, 256, 0, stream>>>(bbox, Wpos[li], bpos[li], aff);
    // attention + residual + relu
    if (li == 0) {
      attn_kernel<<<dim3(256, 16), 32, 0, stream>>>(qbf, kbf, kwt, aff, xfc, bo[li],
                                                    nullptr, actB);
      // fc2: x3 = x2 @ Wfc2 + b
      transpose_cvt<<<dim3(32, 32), tb32x8, 0, stream>>>(Wfc2, bfW, 1024, 1024);
      gemm_bf16<<<dim3(32, 8, 1), 256, 0, stream>>>(actB, bfW, bfc2, xfc, actA, nullptr,
          4096, 1024, 1024, 1024, 1024, 1024, 0, 0L, 0L);
      xin = actA;
    } else {
      attn_kernel<<<dim3(256, 16), 32, 0, stream>>>(qbf, kbf, kwt, aff, xfc, bo[li],
                                                    (float*)d_out, nullptr);
    }
  }
}
